// SelfAttention_64072322122239
// MI455X (gfx1250) — matrix operand order repaired and checked by reading, not compiled, not run on hardware
//
#include <hip/hip_runtime.h>

// ---------------------------------------------------------------------------
// Self-attention for MI455X (gfx1250, wave32, WMMA, async global->LDS copies).
// Pipeline (all bf16 WMMA with f32 accumulate):
//   1. Qp  = query @ Wq^T + bq          (bf16, ws)           [convert-in-stage]
//   2. Kp  = key   @ Wk^T + bk          (bf16, ws)           [convert-in-stage]
//   3. Vpt = (value @ Wv^T + bv)^T      (bf16, ws, [n][v][k])[convert-in-stage]
//   4. S   = Qp @ Kp^T, mask -> NEG     (f32, ws)            [async double-buf]
//   5. P   = softmax(S / 32) rows       (bf16, ws)
//   6. out = P @ Vpt^T                  (f32, d_out)         [async double-buf]
// ---------------------------------------------------------------------------

#define NEGV (-1000000000.0f)

typedef __attribute__((ext_vector_type(16))) __bf16 v16bf;
typedef __attribute__((ext_vector_type(8)))  float  v8f;

struct __align__(16) U128 { unsigned x, y, z, w; };
struct __align__(16) U256 { U128 lo, hi; };
struct __align__(16) F128 { float x, y, z, w; };
struct __align__(8)  U64p { unsigned x, y; };

__device__ __forceinline__ unsigned short f2bf(float f) {
  unsigned u = __builtin_bit_cast(unsigned, f);
  u += 0x7fffu + ((u >> 16) & 1u);          // round-to-nearest-even
  return (unsigned short)(u >> 16);
}

// Packed f32x2 -> bf16x2 conversion (single VALU op, RNE).
__device__ __forceinline__ unsigned cvt_pk_bf16(float lo, float hi) {
  unsigned r;
  asm("v_cvt_pk_bf16_f32 %0, %1, %2" : "=v"(r) : "v"(lo), "v"(hi));
  return r;
}

// LDS byte address for async-copy operands: flat shared pointers truncate to
// the wave-relative LDS offset in the low 32 bits (ISA 10.2 aperture rules).
__device__ __forceinline__ unsigned lds_addr(const void* p) {
  return (unsigned)(unsigned long long)p;
}

// GLOBAL_LOAD_ASYNC_TO_LDS_B128: per-lane 16B copy global -> LDS, ASYNCcnt.
__device__ __forceinline__ void async_cp_b128(unsigned lds, const void* g) {
  asm volatile("global_load_async_to_lds_b128 %0, %1, off"
               :: "v"(lds), "v"(g) : "memory");
}
__device__ __forceinline__ void wait_async_le8() {
  asm volatile("s_wait_asynccnt 0x8" ::: "memory");
}
__device__ __forceinline__ void wait_async_le0() {
  asm volatile("s_wait_asynccnt 0x0" ::: "memory");
}

static constexpr int MBLK = 128;   // tile rows per workgroup
static constexpr int NBLK = 128;   // tile cols per workgroup
static constexpr int KB   = 64;    // staged contraction depth
static constexpr int LDSS = 72;    // KB + 8 pad (elements) to spread LDS banks

// MODE: 0 = bf16 store (+bias), 1 = bf16 transposed store (+bias),
//       2 = f32 store with mask->NEG,   3 = plain f32 store
template<bool SRC_F32, int MODE>
__global__ __launch_bounds__(256)
void gemm_bt(const void* __restrict__ Ag, const void* __restrict__ Bg,
             const float* __restrict__ bias, void* __restrict__ Cg,
             const unsigned char* __restrict__ Mg,
             int M, int Nc, int Kc,
             long long sA, long long sB, long long sC, long long sM, int ldc)
{
  (void)M; (void)Nc;
  const int z    = blockIdx.z;
  const int m0   = blockIdx.y * MBLK;
  const int n0   = blockIdx.x * NBLK;
  const int tid  = threadIdx.x;
  const int lane = tid & 31;
  const int wave = tid >> 5;
  const int wm   = wave >> 1;       // 0..3 -> M sub-block of 32
  const int wn   = wave & 1;        // 0..1 -> N sub-block of 64
  const int lh   = lane >> 4;       // lane half selects K slice
  const int l15  = lane & 15;

  v8f acc[2][4];
#pragma unroll
  for (int i = 0; i < 2; ++i)
#pragma unroll
    for (int j = 0; j < 4; ++j)
#pragma unroll
      for (int e = 0; e < 8; ++e) acc[i][j][e] = 0.0f;

  // 16 WMMAs on one staged k-slab (two k-chunks of 32).
  auto mma_slab = [&](const unsigned short* Ab, const unsigned short* Bb) {
#pragma unroll
    for (int kc = 0; kc < 2; ++kc) {
      const int ko = kc * 32;
      v16bf afr[2], bfr[4];
      // A fragment 16x32: lanes 0-15 -> K 0..15, 16-31 -> K 16..31
#pragma unroll
      for (int mi = 0; mi < 2; ++mi) {
        const int row = wm * 32 + mi * 16 + l15;
        const U128* p = (const U128*)&Ab[row * LDSS + ko + lh * 16];
        U256 t; t.lo = p[0]; t.hi = p[1];
        afr[mi] = __builtin_bit_cast(v16bf, t);
      }
      // B fragment 32x16 (column-major LDS): lane half -> K 0..15 / 16..31
#pragma unroll
      for (int ni = 0; ni < 4; ++ni) {
        const int col = wn * 64 + ni * 16 + l15;
        const U128* p = (const U128*)&Bb[col * LDSS + ko + lh * 16];
        U256 t; t.lo = p[0]; t.hi = p[1];
        bfr[ni] = __builtin_bit_cast(v16bf, t);
      }
#pragma unroll
      for (int mi = 0; mi < 2; ++mi)
#pragma unroll
        for (int ni = 0; ni < 4; ++ni)
          acc[mi][ni] = __builtin_amdgcn_wmma_f32_16x16x32_bf16(
              false, afr[mi], false, bfr[ni], (short)0, acc[mi][ni], false, false);
    }
  };

  if constexpr (!SRC_F32) {
    // bf16 sources: async global->LDS copies, double-buffered pipeline.
    __shared__ __align__(16) unsigned short As[2][MBLK * LDSS];
    __shared__ __align__(16) unsigned short Bs[2][NBLK * LDSS];
    const unsigned short* A = (const unsigned short*)Ag + (long long)z * sA;
    const unsigned short* B = (const unsigned short*)Bg + (long long)z * sB;
    const int c8 = tid & 7;           // 16B chunk within a 64-elem row
    const int r0 = tid >> 3;          // base row (stride 32 per iteration)

    auto stage = [&](int buf, int kb) {   // 8 async b128 copies per wave
#pragma unroll
      for (int i = 0; i < 4; ++i) {
        const int row = r0 + i * 32;
        async_cp_b128(lds_addr(&As[buf][row * LDSS + c8 * 8]),
                      A + (long long)(m0 + row) * Kc + kb + c8 * 8);
        async_cp_b128(lds_addr(&Bs[buf][row * LDSS + c8 * 8]),
                      B + (long long)(n0 + row) * Kc + kb + c8 * 8);
      }
    };

    stage(0, 0);
    for (int kb = 0; kb < Kc; kb += KB) {
      const int cur = (kb / KB) & 1;
      const bool has_next = (kb + KB) < Kc;
      if (has_next) {
        stage(cur ^ 1, kb + KB);      // keep next slab in flight
        wait_async_le8();             // oldest 8 (= current slab) have landed
      } else {
        wait_async_le0();
      }
      __syncthreads();                // all waves' copies visible
      mma_slab(&As[cur][0], &Bs[cur][0]);
      __syncthreads();                // done reading before overwrite
    }
  } else {
    // f32 sources: convert to bf16 while staging (projection GEMMs).
    __shared__ __align__(16) unsigned short As[MBLK * LDSS];
    __shared__ __align__(16) unsigned short Bs[NBLK * LDSS];
    const float* A = (const float*)Ag + (long long)z * sA;
    const float* B = (const float*)Bg + (long long)z * sB;
    for (int kb = 0; kb < Kc; kb += KB) {
#pragma unroll
      for (int i = 0; i < 8; ++i) {
        const int idx = tid + i * 256;       // 2048 float4 chunks per matrix
        const int row = idx >> 4;
        const int c4  = idx & 15;
        F128 fa = *(const F128*)(A + (long long)(m0 + row) * Kc + kb + c4 * 4);
        U64p pa;
        pa.x = cvt_pk_bf16(fa.x, fa.y);
        pa.y = cvt_pk_bf16(fa.z, fa.w);
        *(U64p*)&As[row * LDSS + c4 * 4] = pa;
        F128 fb = *(const F128*)(B + (long long)(n0 + row) * Kc + kb + c4 * 4);
        U64p pb;
        pb.x = cvt_pk_bf16(fb.x, fb.y);
        pb.y = cvt_pk_bf16(fb.z, fb.w);
        *(U64p*)&Bs[row * LDSS + c4 * 4] = pb;
      }
      __syncthreads();
      mma_slab(As, Bs);
      __syncthreads();
    }
  }

  // Epilogue. C fragment: lane l, vgpr r -> row = r + 8*(l>=16), col = l&15.
  unsigned short* Cb = (unsigned short*)Cg + (long long)z * sC;
  float*          Cf = (float*)Cg + (long long)z * sC;
  const unsigned char* mb = (MODE == 2) ? (Mg + (long long)z * sM) : nullptr;

#pragma unroll
  for (int mi = 0; mi < 2; ++mi) {
#pragma unroll
    for (int ni = 0; ni < 4; ++ni) {
      const int gcol = n0 + wn * 64 + ni * 16 + l15;
      const float bv = (MODE <= 1) ? bias[gcol] : 0.0f;
#pragma unroll
      for (int r = 0; r < 8; ++r) {
        const int grow = m0 + wm * 32 + mi * 16 + lh * 8 + r;
        const float v = acc[mi][ni][r] + bv;
        if (MODE == 0) {
          Cb[(long long)grow * ldc + gcol] = f2bf(v);
        } else if (MODE == 1) {
          Cb[(long long)gcol * ldc + grow] = f2bf(v);     // transposed (Vpt)
        } else if (MODE == 2) {
          Cf[(long long)grow * ldc + gcol] =
              mb[(long long)grow * ldc + gcol] ? NEGV : v; // mask BEFORE scale
        } else {
          Cf[(long long)grow * ldc + gcol] = v;
        }
      }
    }
  }
}

// Row softmax: one 256-thread block per 2048-col row; scale 1/sqrt(1024)=1/32.
__global__ __launch_bounds__(256)
void softmax_rows(const float* __restrict__ S, unsigned short* __restrict__ P,
                  int Kcols)
{
  const long long row = blockIdx.x;
  const float* s = S + row * Kcols;
  unsigned short* p = P + row * Kcols;
  const int tid  = threadIdx.x;
  const int lane = tid & 31;
  const int wv   = tid >> 5;

  __align__(16) float a[8];
  *(F128*)&a[0] = *(const F128*)(s + (long long)tid * 4);
  *(F128*)&a[4] = *(const F128*)(s + ((long long)tid + 256) * 4);

  float m = -3.4e38f;
#pragma unroll
  for (int i = 0; i < 8; ++i) m = fmaxf(m, a[i]);
#pragma unroll
  for (int off = 16; off > 0; off >>= 1)
    m = fmaxf(m, __shfl_xor(m, off, 32));

  __shared__ float red[8];
  if (lane == 0) red[wv] = m;
  __syncthreads();
  float rowmax = red[0];
#pragma unroll
  for (int w = 1; w < 8; ++w) rowmax = fmaxf(rowmax, red[w]);

  float sum = 0.0f;
#pragma unroll
  for (int i = 0; i < 8; ++i) {
    a[i] = __expf((a[i] - rowmax) * 0.03125f);   // /sqrt(DIM_Q)
    sum += a[i];
  }
#pragma unroll
  for (int off = 16; off > 0; off >>= 1)
    sum += __shfl_xor(sum, off, 32);

  __syncthreads();                    // red reads above complete
  if (lane == 0) red[wv] = sum;
  __syncthreads();
  float tot = 0.0f;
#pragma unroll
  for (int w = 0; w < 8; ++w) tot += red[w];
  const float inv = 1.0f / tot;

  U64p o0, o1;
  o0.x = cvt_pk_bf16(a[0] * inv, a[1] * inv);
  o0.y = cvt_pk_bf16(a[2] * inv, a[3] * inv);
  o1.x = cvt_pk_bf16(a[4] * inv, a[5] * inv);
  o1.y = cvt_pk_bf16(a[6] * inv, a[7] * inv);
  *(U64p*)(p + (long long)tid * 4) = o0;
  *(U64p*)(p + ((long long)tid + 256) * 4) = o1;
}

extern "C" void kernel_launch(void* const* d_in, const int* in_sizes, int n_in,
                              void* d_out, int out_size, void* d_ws, size_t ws_size,
                              hipStream_t stream) {
  (void)in_sizes; (void)n_in; (void)out_size; (void)ws_size;
  const int N = 8, K = 2048, D = 1024;   // DIM_IN = DIM_Q = DIM_V = 1024

  const float*         query = (const float*)d_in[0];
  const float*         key   = (const float*)d_in[1];
  const float*         value = (const float*)d_in[2];
  const unsigned char* mask  = (const unsigned char*)d_in[3];
  const float* Wq = (const float*)d_in[4];
  const float* bq = (const float*)d_in[5];
  const float* Wk = (const float*)d_in[6];
  const float* bk = (const float*)d_in[7];
  const float* Wv = (const float*)d_in[8];
  const float* bv = (const float*)d_in[9];
  float* out = (float*)d_out;

  // Workspace layout (bytes):
  //   Qp 32MB | Kp 32MB | Vpt 32MB | S 134MB | P 67MB   (~297MB total)
  char* ws = (char*)d_ws;
  const long long szProj = (long long)N * K * D * 2;      // 33554432
  unsigned short* Qp  = (unsigned short*)(ws);
  unsigned short* Kp  = (unsigned short*)(ws + szProj);
  unsigned short* Vpt = (unsigned short*)(ws + 2 * szProj);
  float*          S   = (float*)(ws + 3 * szProj);
  unsigned short* P   = (unsigned short*)(ws + 3 * szProj + (long long)N * K * K * 4);

  const dim3 blk(256);
  const long long sIn  = (long long)K * D;   // per-batch stride of (K,D) tensors
  const long long sKK  = (long long)K * K;   // per-batch stride of (K,K) tensors

  // 1..2: Q/K projections  (A f32 (K,D), B = W (D,D) row-major = B^T pattern)
  const dim3 gproj(D / NBLK, K / MBLK, N);
  gemm_bt<true, 0><<<gproj, blk, 0, stream>>>(
      query, Wq, bq, Qp, nullptr, K, D, D, sIn, 0, sIn, 0, D);
  gemm_bt<true, 0><<<gproj, blk, 0, stream>>>(
      key, Wk, bk, Kp, nullptr, K, D, D, sIn, 0, sIn, 0, D);
  // 3: V projection, stored transposed as Vpt[n][v][k]  (ldc = K)
  gemm_bt<true, 1><<<gproj, blk, 0, stream>>>(
      value, Wv, bv, Vpt, nullptr, K, D, D, sIn, 0, sIn, 0, K);

  // 4: scores S = Qp @ Kp^T with mask -> NEG (before /32 scaling)
  const dim3 gsc(K / NBLK, K / MBLK, N);
  gemm_bt<false, 2><<<gsc, blk, 0, stream>>>(
      Qp, Kp, nullptr, S, mask, K, K, D, sIn, sIn, sKK, sKK, K);

  // 5: row softmax (scale 1/32 inside), P bf16
  softmax_rows<<<dim3(N * K), blk, 0, stream>>>(S, P, K);

  // 6: out = P @ Vpt^T  (contraction over keys, length K)
  const dim3 gf(D / NBLK, K / MBLK, N);
  gemm_bt<false, 3><<<gf, blk, 0, stream>>>(
      P, Vpt, nullptr, out, nullptr, K, D, K, sKK, (long long)D * K, sIn, 0, D);
}
